// SPDDA_28784870818047
// MI455X (gfx1250) — compile-verified, matrix operand order checked by reading, not run on hardware
//
#include <hip/hip_runtime.h>

// ---------------- problem constants ----------------
#define Bn   8
#define Cn   128
#define Hn   96
#define Wd   96
#define HWn  (Hn*Wd)          // 9216
#define NHn  2
#define CHn  (Cn/NHn)         // 64
#define Kn   96               // 3*C/4 selected channels
#define Rn   10
#define KSn  21
#define NT_N (HWn/16)         // 576 N-tiles per image

typedef __attribute__((ext_vector_type(16))) __bf16        v16bf;
typedef __attribute__((ext_vector_type(8)))  float         v8f;
typedef __attribute__((ext_vector_type(4)))  float         v4f;
typedef __attribute__((ext_vector_type(4)))  unsigned int  v4u;
typedef __attribute__((ext_vector_type(8)))  unsigned int  v8u;

__device__ __forceinline__ float gelu_f(float x) {
  return 0.5f * x * (1.0f + erff(x * 0.70710678118654752f));
}

__device__ __forceinline__ v8f wmma_bf16(v16bf a, v16bf b, v8f c) {
  // D(f32,16x16) = A(bf16,16x32) * B(bf16,32x16) + C
  return __builtin_amdgcn_wmma_f32_16x16x32_bf16(false, a, false, b, (short)0, c,
                                                 false, false);
}

// Fragment loader: K contiguous in memory. Lane's row pointer advanced to k0.
// VGPR v holds K-pair 16*(v/4) + 8*hi + 2*(v%4)  ->  two 16B loads per lane.
__device__ __forceinline__ v16bf load_frag(const __bf16* __restrict__ rowk0, int hi) {
  v4u a = *(const v4u*)(rowk0 + (hi << 3));
  v4u b = *(const v4u*)(rowk0 + 16 + (hi << 3));
  v8u u = __builtin_shufflevector(a, b, 0, 1, 2, 3, 4, 5, 6, 7);
  return __builtin_bit_cast(v16bf, u);
}

// Same, but branchlessly zeroed when !valid (loads are address-clamped safe).
__device__ __forceinline__ v16bf load_frag_masked(const __bf16* __restrict__ rowk0,
                                                  int hi, bool valid) {
  v4u a = *(const v4u*)(rowk0 + (hi << 3));
  v4u b = *(const v4u*)(rowk0 + 16 + (hi << 3));
  unsigned mk = valid ? 0xFFFFFFFFu : 0u;
  v4u mv = {mk, mk, mk, mk};
  a &= mv; b &= mv;
  v8u u = __builtin_shufflevector(a, b, 0, 1, 2, 3, 4, 5, 6, 7);
  return __builtin_bit_cast(v16bf, u);
}

// ---------------- converts / repacks ----------------
__global__ void cvt_bf16_kernel(const float* __restrict__ in, __bf16* __restrict__ out, int n) {
  int i = blockIdx.x * blockDim.x + threadIdx.x;
  if (i < n) out[i] = (__bf16)in[i];
}

// NCHW f32 -> channel-last bf16  [b][p][c]
__global__ void cvt_chlast_kernel(const float* __restrict__ in, __bf16* __restrict__ out) {
  int idx = blockIdx.x * blockDim.x + threadIdx.x;      // (b*HW + p)*C + c
  if (idx >= Bn * Cn * HWn) return;
  int c = idx & (Cn - 1);
  int bp = idx >> 7;
  int p = bp % HWn, b = bp / HWn;
  out[idx] = (__bf16)in[((size_t)b * Cn + c) * HWn + p];
}

// enc weights OIHW f32 -> [tap][co][ci] bf16 (K contiguous per tap)
__global__ void repack_encw_kernel(const float* __restrict__ in, __bf16* __restrict__ out) {
  int idx = blockIdx.x * blockDim.x + threadIdx.x;      // (t*Cn + co)*Cn + ci
  if (idx >= 9 * Cn * Cn) return;
  int ci = idx & (Cn - 1);
  int rest = idx >> 7;
  int co = rest & (Cn - 1);
  int t  = rest >> 7;
  out[idx] = (__bf16)in[((size_t)co * Cn + ci) * 9 + t];
}

// ---------------- encoder 3x3 conv, implicit GEMM (WMMA) ----------------
// Wt: [t][co][ci] bf16, Xb: [b][p][c] bf16, Out: [b][p][c] f32 (channel-last)
__global__ __launch_bounds__(128)
void enc_conv_kernel(const __bf16* __restrict__ Wt, const __bf16* __restrict__ Xb,
                     float* __restrict__ Out, const float* __restrict__ bias) {
  int wave = threadIdx.x >> 5, lane = threadIdx.x & 31;
  int tile = blockIdx.x * 4 + wave;
  int nt = tile % NT_N;
  int rest = tile / NT_N;
  int mt = rest & 7;            // Cn/16 = 8
  int b  = rest >> 3;
  if (b >= Bn) return;
  int lo = lane & 15, hi = lane >> 4;
  int p0 = nt * 16, co0 = mt * 16;
  int p  = p0 + lo, py = p / Wd, px = p % Wd;

  v8f acc = {};
  #pragma unroll
  for (int t = 0; t < 9; ++t) {
    int yy = py + t / 3 - 1, xx = px + t % 3 - 1;
    bool valid = (yy >= 0) & (yy < Hn) & (xx >= 0) & (xx < Wd);
    int psrc = valid ? (yy * Wd + xx) : 0;
    const __bf16* bbase = Xb + ((size_t)b * HWn + psrc) * Cn;            // K=ci contiguous
    const __bf16* abase = Wt + ((size_t)t * Cn + co0 + lo) * Cn;         // K=ci contiguous
    #pragma unroll
    for (int k0 = 0; k0 < Cn; k0 += 32) {
      v16bf a  = load_frag(abase + k0, hi);
      v16bf bm = load_frag_masked(bbase + k0, hi, valid);
      acc = wmma_bf16(a, bm, acc);
    }
  }
  // channel-last store: 8 consecutive channels per half-wave, two 16B stores
  float* op = Out + ((size_t)b * HWn + p) * Cn + co0 + (hi << 3);
  v4f r0, r1;
  #pragma unroll
  for (int v = 0; v < 4; ++v) r0[v] = acc[v] + bias[co0 + (hi << 3) + v];
  #pragma unroll
  for (int v = 0; v < 4; ++v) r1[v] = acc[4 + v] + bias[co0 + (hi << 3) + 4 + v];
  *(v4f*)op = r0;
  *(v4f*)(op + 4) = r1;
}

// ---------------- generic 1x1 conv GEMM (WMMA) ---------------------------
// Wb: [co][ci] bf16, In: [b][p][ci] bf16 (channel-last)
// Out: channel-last [b][p][co] if chFirst==0, else channel-first [b][co][p]
__global__ __launch_bounds__(128)
void gemm1x1_kernel(const __bf16* __restrict__ Wb, const __bf16* __restrict__ In,
                    float* __restrict__ Out, int Cin, int applyGelu, int chFirst) {
  int wave = threadIdx.x >> 5, lane = threadIdx.x & 31;
  int tile = blockIdx.x * 4 + wave;
  int nt = tile % NT_N;
  int rest = tile / NT_N;
  int mt = rest & 7;            // Cout = Cn always
  int b  = rest >> 3;
  if (b >= Bn) return;
  int lo = lane & 15, hi = lane >> 4;
  int p0 = nt * 16, co0 = mt * 16;
  int p  = p0 + lo;

  v8f acc = {};
  const __bf16* arow = Wb + (size_t)(co0 + lo) * Cin;
  const __bf16* brow = In + ((size_t)b * HWn + p) * Cin;
  for (int k0 = 0; k0 < Cin; k0 += 32) {
    if (k0 + 32 < Cin) __builtin_prefetch(brow + k0 + 32, 0, 1);
    v16bf a  = load_frag(arow + k0, hi);
    v16bf bm = load_frag(brow + k0, hi);
    acc = wmma_bf16(a, bm, acc);
  }
  if (chFirst) {
    float* outp = Out + (size_t)b * Cn * HWn + p;
    #pragma unroll
    for (int v = 0; v < 8; ++v) {
      int m = v + (hi << 3);
      float val = acc[v];
      if (applyGelu) val = gelu_f(val);
      outp[(size_t)(co0 + m) * HWn] = val;
    }
  } else {
    float* op = Out + ((size_t)b * HWn + p) * Cn + co0 + (hi << 3);
    v4f r0, r1;
    #pragma unroll
    for (int v = 0; v < 4; ++v) r0[v] = applyGelu ? gelu_f(acc[v]) : acc[v];
    #pragma unroll
    for (int v = 0; v < 4; ++v) r1[v] = applyGelu ? gelu_f(acc[4 + v]) : acc[4 + v];
    *(v4f*)op = r0;
    *(v4f*)(op + 4) = r1;
  }
}

// ---------------- channel LayerNorm (contiguous rows), emits bf16 --------
__global__ void chan_ln_kernel(const float* __restrict__ in, const float* __restrict__ g,
                               const float* __restrict__ bt, __bf16* __restrict__ outb) {
  int idx = blockIdx.x * blockDim.x + threadIdx.x;     // over B*HW pixels
  if (idx >= Bn * HWn) return;
  const float* row = in + (size_t)idx * Cn;
  float m = 0.f;
  for (int c = 0; c < Cn; ++c) m += row[c];
  m *= (1.0f / Cn);
  float v = 0.f;
  for (int c = 0; c < Cn; ++c) { float d = row[c] - m; v += d * d; }
  v *= (1.0f / Cn);
  float inv = rsqrtf(v + 1e-5f);
  __bf16* ob = outb + (size_t)idx * Cn;
  for (int c = 0; c < Cn; ++c) ob[c] = (__bf16)((row[c] - m) * inv * g[c] + bt[c]);
}

// ---------------- depthwise 3x3 conv, channel-last in/out ----------------
__global__ void dwconv_kernel(const float* __restrict__ in, const float* __restrict__ w,
                              float* __restrict__ out) {
  int idx = blockIdx.x * blockDim.x + threadIdx.x;     // (b*HW + p)*C + c
  if (idx >= Bn * Cn * HWn) return;
  int c = idx & (Cn - 1);
  int bp = idx >> 7;
  int p = bp % HWn, b = bp / HWn;
  int py = p / Wd, px = p % Wd;
  const float* base = in + (size_t)b * HWn * Cn + c;
  float acc = 0.f;
  #pragma unroll
  for (int t = 0; t < 9; ++t) {
    int yy = py + t / 3 - 1, xx = px + t % 3 - 1;
    if (yy >= 0 && yy < Hn && xx >= 0 && xx < Wd)
      acc += w[c * 9 + t] * base[(size_t)(yy * Wd + xx) * Cn];
  }
  out[idx] = acc;
}

// ---------------- L2 row-norm + transpose to channel-first bf16 ----------
// in: channel-last [b][p][c]; out rows: [b*C + ch][HW] (contiguous writes)
__global__ __launch_bounds__(256)
void rownorm_kernel(const float* __restrict__ in, __bf16* __restrict__ outb) {
  int row = blockIdx.x;                                 // B*C rows
  int tid = threadIdx.x;
  int b = row >> 7, ch = row & (Cn - 1);
  const float* src = in + (size_t)b * HWn * Cn + ch;
  __shared__ float red[256];
  float s = 0.f;
  for (int i = tid; i < HWn; i += 256) { float x = src[(size_t)i * Cn]; s += x * x; }
  red[tid] = s; __syncthreads();
  for (int off = 128; off; off >>= 1) {
    if (tid < off) red[tid] += red[tid + off];
    __syncthreads();
  }
  float inv = 1.0f / fmaxf(sqrtf(red[0]), 1e-12f);
  __bf16* ob = outb + (size_t)row * HWn;
  for (int i = tid; i < HWn; i += 256) ob[i] = (__bf16)(src[(size_t)i * Cn] * inv);
}

// ---------------- attention score GEMM: attn = (q k^T) * scale ------------
// Qb/Kb: channel-first rows [bh*64 + c][HW]
__global__ __launch_bounds__(128)
void attn_gemm_kernel(const __bf16* __restrict__ Qb, const __bf16* __restrict__ Kb,
                      float* __restrict__ Attn, const float* __restrict__ scale) {
  int wave = threadIdx.x >> 5, lane = threadIdx.x & 31;
  int tile = blockIdx.x * 4 + wave;                     // 256 tiles
  int nt = tile & 3, mt = (tile >> 2) & 3, bh = tile >> 4;
  if (bh >= Bn * NHn) return;
  int lo = lane & 15, hi = lane >> 4;
  const __bf16* qrow = Qb + ((size_t)bh * CHn + mt * 16 + lo) * HWn;  // A[m][k]
  const __bf16* krow = Kb + ((size_t)bh * CHn + nt * 16 + lo) * HWn;  // B[k][n]=K[n][k]
  v8f acc = {};
  for (int k0 = 0; k0 < HWn; k0 += 32) {
    if (k0 + 32 < HWn) __builtin_prefetch(qrow + k0 + 32, 0, 1);
    v16bf a  = load_frag(qrow + k0, hi);
    v16bf bm = load_frag(krow + k0, hi);
    acc = wmma_bf16(a, bm, acc);
  }
  float sc = scale[bh % NHn];
  #pragma unroll
  for (int v = 0; v < 8; ++v) {
    int m = v + (hi << 3);
    Attn[((size_t)bh * CHn + mt * 16 + m) * CHn + nt * 16 + lo] = acc[v] * sc;
  }
}

// ---------------- softmax rows of 64 + per-channel sigma ------------------
__global__ void softmax_sig_kernel(const float* __restrict__ attn,
                                   __bf16* __restrict__ smb, float* __restrict__ sig) {
  int row = blockIdx.x * blockDim.x + threadIdx.x;      // B*C rows
  if (row >= Bn * Cn) return;
  const float* a = attn + (size_t)row * CHn;
  float mean = 0.f;
  for (int j = 0; j < CHn; ++j) mean += a[j];
  mean *= (1.0f / CHn);
  float var = 0.f;
  for (int j = 0; j < CHn; ++j) { float d = a[j] - mean; var += d * d; }
  var *= (1.0f / (CHn - 1));                            // ddof=1
  sig[row] = sqrtf(var + 1e-8f);
  float mx = -3.0e38f;
  for (int j = 0; j < CHn; ++j) mx = fmaxf(mx, a[j]);
  float s = 0.f;
  for (int j = 0; j < CHn; ++j) s += expf(a[j] - mx);
  float invs = 1.0f / s;
  __bf16* ob = smb + (size_t)row * CHn;
  for (int j = 0; j < CHn; ++j) ob[j] = (__bf16)(expf(a[j] - mx) * invs);
}

// ---------------- attention output GEMM: out = softmax @ v ---------------
// Smb: [bh*64 + c][64]; Vb channel-last [b][p][c]; Out channel-last [b][p][c]
__global__ __launch_bounds__(128)
void attn_out_gemm_kernel(const __bf16* __restrict__ Smb, const __bf16* __restrict__ Vb,
                          float* __restrict__ Out) {
  int wave = threadIdx.x >> 5, lane = threadIdx.x & 31;
  int tile = blockIdx.x * 4 + wave;
  int nt = tile % NT_N;
  int rest = tile / NT_N;
  int mt = rest & 3;                                    // 64/16
  int bh = rest >> 2;
  if (bh >= Bn * NHn) return;
  int h = bh & (NHn - 1), b = bh >> 1;
  int lo = lane & 15, hi = lane >> 4;
  int p = nt * 16 + lo;
  const __bf16* arow = Smb + ((size_t)bh * CHn + mt * 16 + lo) * CHn;       // A[m][k]
  const __bf16* brow = Vb + ((size_t)b * HWn + p) * Cn + h * CHn;           // B[k][n]
  v8f acc = {};
  #pragma unroll
  for (int k0 = 0; k0 < CHn; k0 += 32) {
    v16bf a  = load_frag(arow + k0, hi);
    v16bf bm = load_frag(brow + k0, hi);
    acc = wmma_bf16(a, bm, acc);
  }
  // channel-last store of 8 consecutive channels
  float* op = Out + ((size_t)b * HWn + p) * Cn + h * CHn + mt * 16 + (hi << 3);
  v4f r0 = {acc[0], acc[1], acc[2], acc[3]};
  v4f r1 = {acc[4], acc[5], acc[6], acc[7]};
  *(v4f*)op = r0;
  *(v4f*)(op + 4) = r1;
}

// ---------------- build gelu(concat[o1,o2]) in bf16 (channel-last) -------
__global__ void cat_gelu_kernel(const float* __restrict__ o1, const float* __restrict__ o2,
                                __bf16* __restrict__ cat) {
  int idx = blockIdx.x * blockDim.x + threadIdx.x;      // (b*HW + p)*C + c
  if (idx >= Bn * Cn * HWn) return;
  int c = idx & (Cn - 1);
  int bp = idx >> 7;
  __bf16* crow = cat + (size_t)bp * (2 * Cn);
  crow[c]      = (__bf16)gelu_f(o1[idx]);
  crow[Cn + c] = (__bf16)gelu_f(o2[idx]);
}

// ---------------- mean/max pooling over HW per (b,c) (channel-first in) --
__global__ __launch_bounds__(256)
void pool_kernel(const float* __restrict__ in, float* __restrict__ pmean,
                 float* __restrict__ pmax) {
  int row = blockIdx.x;                                 // B*C
  int tid = threadIdx.x;
  const float* r = in + (size_t)row * HWn;
  __shared__ float rs[256], rm[256];
  float s = 0.f, m = -3.0e38f;
  for (int i = tid; i < HWn; i += 256) { float x = r[i]; s += x; m = fmaxf(m, x); }
  rs[tid] = s; rm[tid] = m; __syncthreads();
  for (int off = 128; off; off >>= 1) {
    if (tid < off) { rs[tid] += rs[tid + off]; rm[tid] = fmaxf(rm[tid], rm[tid + off]); }
    __syncthreads();
  }
  if (tid == 0) { pmean[row] = rs[0] * (1.0f / HWn); pmax[row] = rm[0]; }
}

// ---------------- channel gate MLP: sigmoid(mlp(mean)+mlp(max)) ----------
__global__ __launch_bounds__(128)
void gate_kernel(const float* __restrict__ pmean, const float* __restrict__ pmax,
                 const float* __restrict__ w1, const float* __restrict__ b1,
                 const float* __restrict__ w2, const float* __restrict__ b2,
                 float* __restrict__ score) {
  int b = blockIdx.x, tid = threadIdx.x;
  __shared__ float pm[Cn], px[Cn], hm[8], hx[8];
  pm[tid] = pmean[b * Cn + tid];
  px[tid] = pmax[b * Cn + tid];
  __syncthreads();
  if (tid < 8) {
    float s = b1[tid];
    for (int c = 0; c < Cn; ++c) s += w1[tid * Cn + c] * pm[c];
    hm[tid] = fmaxf(s, 0.f);
  } else if (tid < 16) {
    int j = tid - 8;
    float s = b1[j];
    for (int c = 0; c < Cn; ++c) s += w1[j * Cn + c] * px[c];
    hx[j] = fmaxf(s, 0.f);
  }
  __syncthreads();
  float s1 = b2[tid], s2 = b2[tid];
  for (int j = 0; j < 8; ++j) { s1 += hm[j] * w2[tid * 8 + j]; s2 += hx[j] * w2[tid * 8 + j]; }
  score[b * Cn + tid] = 1.0f / (1.0f + expf(-(s1 + s2)));
}

// ---------------- top-K channel select, ascending-sorted indices ----------
__global__ __launch_bounds__(128)
void topk_kernel(const float* __restrict__ score, int* __restrict__ kidx) {
  int b = blockIdx.x, tid = threadIdx.x;
  __shared__ float sc[Cn];
  __shared__ int   sel[Cn];
  sc[tid] = score[b * Cn + tid];
  __syncthreads();
  int beats = 0;
  for (int c = 0; c < Cn; ++c)
    if (sc[c] > sc[tid] || (sc[c] == sc[tid] && c < tid)) beats++;
  sel[tid] = (beats < Kn) ? 1 : 0;
  __syncthreads();
  int slot = 0;
  for (int c = 0; c < tid; ++c) slot += sel[c];
  if (sel[tid]) kidx[b * Kn + slot] = tid;
}

// ---------------- gaussian mask + softmax -> mixing weights --------------
__global__ void mixweights_kernel(const int* __restrict__ kidx, const float* __restrict__ sig,
                                  float* __restrict__ wm) {
  int id = blockIdx.x * blockDim.x + threadIdx.x;       // over B*K
  if (id >= Bn * Kn) return;
  int b = id / Kn;
  int c = kidx[id];
  float s = sig[b * Cn + c];
  float m[KSn];
  float mx = -3.0e38f;
  #pragma unroll
  for (int j = 0; j < KSn; ++j) {
    float xv = -5.0f + 0.5f * j;
    float t = xv / s;
    float g = expf(-0.5f * t * t) / (s * 2.5066282746310002f);
    float mv = (fabsf(g) <= s) ? g : 0.0f;
    m[j] = mv;
    mx = fmaxf(mx, mv);
  }
  float sum = 0.f;
  #pragma unroll
  for (int j = 0; j < KSn; ++j) { m[j] = expf(m[j] - mx); sum += m[j]; }
  float inv = 1.0f / sum;
  #pragma unroll
  for (int j = 0; j < KSn; ++j) wm[id * KSn + j] = m[j] * inv;
}

// ---------------- 21-tap adaptive channel mixing (fea channel-last) ------
__global__ void mix_kernel(const float* __restrict__ fea, const int* __restrict__ kidx,
                           const float* __restrict__ wm, float* __restrict__ out) {
  int idx = blockIdx.x * blockDim.x + threadIdx.x;      // over B*K*HW
  if (idx >= Bn * Kn * HWn) return;
  int p = idx % HWn, bi = idx / HWn;
  int i = bi % Kn, b = bi / Kn;
  const float* wr = wm + (size_t)bi * KSn;
  const float* frow = fea + ((size_t)b * HWn + p) * Cn;   // one 512B row, L2-resident
  float acc = 0.f;
  #pragma unroll
  for (int j = 0; j < KSn; ++j) {
    int src = i + j - Rn;
    if (src >= 0 && src < Kn) acc += wr[j] * frow[kidx[b * Kn + src]];
  }
  out[idx] = acc;
}

// =========================================================================
extern "C" void kernel_launch(void* const* d_in, const int* in_sizes, int n_in,
                              void* d_out, int out_size, void* d_ws, size_t ws_size,
                              hipStream_t stream) {
  const float* x      = (const float*)d_in[0];
  const float* enc_w  = (const float*)d_in[1];
  const float* enc_b  = (const float*)d_in[2];
  const float* ln_in_g= (const float*)d_in[3];
  const float* ln_in_b= (const float*)d_in[4];
  const float* qw     = (const float*)d_in[5];
  const float* qdw    = (const float*)d_in[6];
  const float* kw     = (const float*)d_in[7];
  const float* kdw    = (const float*)d_in[8];
  const float* vw     = (const float*)d_in[9];
  const float* vdw    = (const float*)d_in[10];
  const float* scale  = (const float*)d_in[11];
  const float* ln_out_g = (const float*)d_in[12];
  const float* ln_out_b = (const float*)d_in[13];
  const float* f1w    = (const float*)d_in[14];
  const float* f1dw   = (const float*)d_in[15];
  const float* f2w    = (const float*)d_in[16];
  const float* f2dw   = (const float*)d_in[17];
  const float* fow    = (const float*)d_in[18];
  const float* g_w1   = (const float*)d_in[19];
  const float* g_b1   = (const float*)d_in[20];
  const float* g_w2   = (const float*)d_in[21];
  const float* g_b2   = (const float*)d_in[22];
  float* outp = (float*)d_out;

  // ---- workspace carve-up ----
  char* wsp = (char*)d_ws;
  auto alloc = [&](size_t bytes) -> char* {
    char* pp = wsp;
    wsp += (bytes + 255) & ~(size_t)255;
    return pp;
  };
  const size_t NE = (size_t)Bn * Cn * HWn;              // 9,437,184
  __bf16* x_bf   = (__bf16*)alloc(NE * 2);              // later reused as oln_bf
  __bf16* xn_bf  = (__bf16*)alloc(NE * 2);
  __bf16* q_bf   = (__bf16*)alloc(NE * 2);
  __bf16* k_bf   = (__bf16*)alloc(NE * 2);
  __bf16* v_bf   = (__bf16*)alloc(NE * 2);
  __bf16* cat_bf = (__bf16*)alloc(NE * 4);              // 2C channels
  float*  fea    = (float*)alloc(NE * 4);               // channel-last
  float*  t0     = (float*)alloc(NE * 4);               // temp / later ffn_out
  float*  qf     = (float*)alloc(NE * 4);               // later o1
  float*  kf     = (float*)alloc(NE * 4);               // later o2
  float*  vf     = (float*)alloc(NE * 4);               // later attention out
  __bf16* enc_wt = (__bf16*)alloc((size_t)9 * Cn * Cn * 2);
  __bf16* wqb    = (__bf16*)alloc((size_t)Cn * Cn * 2);
  __bf16* wkb    = (__bf16*)alloc((size_t)Cn * Cn * 2);
  __bf16* wvb    = (__bf16*)alloc((size_t)Cn * Cn * 2);
  __bf16* wf1b   = (__bf16*)alloc((size_t)Cn * Cn * 2);
  __bf16* wf2b   = (__bf16*)alloc((size_t)Cn * Cn * 2);
  __bf16* wfob   = (__bf16*)alloc((size_t)Cn * 2 * Cn * 2);
  float*  attn   = (float*)alloc((size_t)Bn * NHn * CHn * CHn * 4);
  __bf16* sm_bf  = (__bf16*)alloc((size_t)Bn * NHn * CHn * CHn * 2);
  float*  sig    = (float*)alloc((size_t)Bn * Cn * 4);
  float*  pmean  = (float*)alloc((size_t)Bn * Cn * 4);
  float*  pmax   = (float*)alloc((size_t)Bn * Cn * 4);
  float*  score  = (float*)alloc((size_t)Bn * Cn * 4);
  int*    kidx   = (int*)alloc((size_t)Bn * Kn * 4);
  float*  wm     = (float*)alloc((size_t)Bn * Kn * KSn * 4);
  float*  aout   = vf;     // attention output reuses vf (channel-last)
  float*  o1f    = qf;
  float*  o2f    = kf;
  float*  ffn    = t0;     // fow output (channel-first) reuses t0
  __bf16* oln_bf = x_bf;   // post-attention LN reuses x_bf

  auto cvt = [&](const float* src, __bf16* dst, int n) {
    cvt_bf16_kernel<<<(n + 255) / 256, 256, 0, stream>>>(src, dst, n);
  };
  const int gemmBlocks = (Bn * (Cn / 16) * NT_N) / 4;   // 9216 blocks, 4 waves each
  const int elemBlocks = (int)((NE + 255) / 256);
  const int pixBlocks  = (Bn * HWn + 255) / 256;

  // 1) layout conversions (bf16, channel-last activations, repacked weights)
  cvt_chlast_kernel<<<elemBlocks, 256, 0, stream>>>(x, x_bf);
  repack_encw_kernel<<<(9 * Cn * Cn + 255) / 256, 256, 0, stream>>>(enc_w, enc_wt);
  cvt(qw, wqb, Cn * Cn);  cvt(kw, wkb, Cn * Cn);  cvt(vw, wvb, Cn * Cn);
  cvt(f1w, wf1b, Cn * Cn); cvt(f2w, wf2b, Cn * Cn); cvt(fow, wfob, Cn * 2 * Cn);

  // 2) encoder 3x3 conv (WMMA implicit GEMM), fea channel-last
  enc_conv_kernel<<<gemmBlocks, 128, 0, stream>>>(enc_wt, x_bf, fea, enc_b);

  // 3) input channel-LN -> bf16 (contiguous rows)
  chan_ln_kernel<<<pixBlocks, 256, 0, stream>>>(fea, ln_in_g, ln_in_b, xn_bf);

  // 4) q / k / v : 1x1 WMMA GEMM + depthwise 3x3 (+ row-norm transpose)
  gemm1x1_kernel<<<gemmBlocks, 128, 0, stream>>>(wqb, xn_bf, t0, Cn, 0, 0);
  dwconv_kernel<<<elemBlocks, 256, 0, stream>>>(t0, qdw, qf);
  rownorm_kernel<<<Bn * Cn, 256, 0, stream>>>(qf, q_bf);

  gemm1x1_kernel<<<gemmBlocks, 128, 0, stream>>>(wkb, xn_bf, t0, Cn, 0, 0);
  dwconv_kernel<<<elemBlocks, 256, 0, stream>>>(t0, kdw, kf);
  rownorm_kernel<<<Bn * Cn, 256, 0, stream>>>(kf, k_bf);

  gemm1x1_kernel<<<gemmBlocks, 128, 0, stream>>>(wvb, xn_bf, t0, Cn, 0, 0);
  dwconv_kernel<<<elemBlocks, 256, 0, stream>>>(t0, vdw, vf);
  cvt(vf, v_bf, (int)NE);                               // stays channel-last

  // 5) attention scores (WMMA, K=9216) + softmax + sigma
  attn_gemm_kernel<<<(Bn * NHn * 16) / 4, 128, 0, stream>>>(q_bf, k_bf, attn, scale);
  softmax_sig_kernel<<<(Bn * Cn + 255) / 256, 256, 0, stream>>>(attn, sm_bf, sig);

  // 6) attention output (WMMA) + output LN
  attn_out_gemm_kernel<<<gemmBlocks, 128, 0, stream>>>(sm_bf, v_bf, aout);
  chan_ln_kernel<<<pixBlocks, 256, 0, stream>>>(aout, ln_out_g, ln_out_b, oln_bf);

  // 7) FFN: two gelu(1x1)->dw branches, gelu(concat) -> 1x1 (K=256)
  gemm1x1_kernel<<<gemmBlocks, 128, 0, stream>>>(wf1b, oln_bf, t0, Cn, 1, 0);
  dwconv_kernel<<<elemBlocks, 256, 0, stream>>>(t0, f1dw, o1f);
  gemm1x1_kernel<<<gemmBlocks, 128, 0, stream>>>(wf2b, oln_bf, t0, Cn, 1, 0);
  dwconv_kernel<<<elemBlocks, 256, 0, stream>>>(t0, f2dw, o2f);
  cat_gelu_kernel<<<elemBlocks, 256, 0, stream>>>(o1f, o2f, cat_bf);
  gemm1x1_kernel<<<gemmBlocks, 128, 0, stream>>>(wfob, cat_bf, ffn, 2 * Cn, 0, 1);

  // 8) channel gate -> score -> top-K sorted indices
  pool_kernel<<<Bn * Cn, 256, 0, stream>>>(ffn, pmean, pmax);
  gate_kernel<<<Bn, 128, 0, stream>>>(pmean, pmax, g_w1, g_b1, g_w2, g_b2, score);
  topk_kernel<<<Bn, 128, 0, stream>>>(score, kidx);

  // 9) gaussian mixing weights + 21-tap adaptive channel mix -> d_out
  mixweights_kernel<<<(Bn * Kn + 255) / 256, 256, 0, stream>>>(kidx, sig, wm);
  mix_kernel<<<(int)(((size_t)Bn * Kn * HWn + 255) / 256), 256, 0, stream>>>(fea, kidx, wm, outp);

  (void)in_sizes; (void)n_in; (void)out_size; (void)ws_size;
}
